// SlotAttention_26396869001330
// MI455X (gfx1250) — matrix operand order repaired and checked by reading, not compiled, not run on hardware
//
#include <hip/hip_runtime.h>

#define N_TOK  524288
#define NBLK_P 1024          // pass-kernel blocks
#define CHUNKS 2             // 256-row chunks per block  (1024*2*256 = N_TOK)
#define EPSV   1e-8f

typedef __attribute__((ext_vector_type(16))) _Float16 v16h;
typedef __attribute__((ext_vector_type(8)))  float    v8f;

union VH { v16h v; uint4 q[2]; };

struct Params {
  const float *x, *noise_fg, *noise_bg, *ln_in_g, *ln_in_b;
  const float *mu_fg, *logsig_fg, *mu_bg, *logsig_bg;
  const float *Wk, *Wv;
  const float *lnq_fg_g, *lnq_fg_b, *Wq_fg, *lnq_bg_g, *lnq_bg_b, *Wq_bg;
  const float *gfg_wih, *gfg_whh, *gfg_bih, *gfg_bhh;
  const float *gbg_wih, *gbg_whh, *gbg_bih, *gbg_bhh;
  const float *mfg_ln_g, *mfg_ln_b, *mfg_w1, *mfg_b1, *mfg_w2, *mfg_b2;
  const float *mbg_ln_g, *mbg_ln_b, *mbg_w1, *mbg_b1, *mbg_w2, *mbg_b2;
  float *out;   // [0..191] slots, [192..192+3N) attn.T
  float *ws;
};

// ws float layout
#define WS_Q    208      // 192 floats: Q' = q @ Wk  (3 x 64, scale folded in)
#define WS_SLOT 400      // 192 floats: current slots
#define WS_PART 1024     // NBLK_P * 208 floats: per-block partials (192 y + 3 den)

// ---------------------------------------------------------------------------
// q = LN(slot) @ Wq.T * 1/sqrt(64)
// ---------------------------------------------------------------------------
__device__ __forceinline__ float compute_q(const Params& p, const float* s_slot,
                                           int s, int d) {
  const float* g  = (s == 0) ? p.lnq_bg_g : p.lnq_fg_g;
  const float* b  = (s == 0) ? p.lnq_bg_b : p.lnq_fg_b;
  const float* Wq = (s == 0) ? p.Wq_bg    : p.Wq_fg;
  const float* sl = s_slot + s * 64;
  float m = 0.f;
  #pragma unroll
  for (int k = 0; k < 64; ++k) m += sl[k];
  m *= (1.f / 64.f);
  float var = 0.f;
  #pragma unroll
  for (int k = 0; k < 64; ++k) { float t = sl[k] - m; var += t * t; }
  var *= (1.f / 64.f);
  float inv = rsqrtf(var + 1e-5f);
  float acc = 0.f;
  #pragma unroll
  for (int k = 0; k < 64; ++k)
    acc += ((sl[k] - m) * inv * g[k] + b[k]) * Wq[d * 64 + k];
  return acc * 0.125f;
}

// Q'[s][k] = sum_d q[s][d] * Wk[d][k]   (collapses the k-GEMM out of the pass)
__device__ __forceinline__ float compute_qp(const Params& p, const float* s_qv,
                                            int s, int k) {
  const float* q = s_qv + s * 64;
  float acc = 0.f;
  #pragma unroll 8
  for (int d = 0; d < 64; ++d) acc += q[d] * p.Wk[d * 64 + k];
  return acc;
}

// ---------------------------------------------------------------------------
// Kernel 0: slot init + first Q'
// ---------------------------------------------------------------------------
__global__ __launch_bounds__(256) void slot_init_k(Params p) {
  __shared__ float s_slot[192];
  __shared__ float s_qv[192];
  const int tid = threadIdx.x;
  if (tid < 192) {
    int s = tid >> 6, d = tid & 63;
    float v = (s == 0)
        ? p.mu_bg[d] + __expf(p.logsig_bg[d]) * p.noise_bg[d]
        : p.mu_fg[d] + __expf(p.logsig_fg[d]) * p.noise_fg[(s - 1) * 64 + d];
    p.ws[WS_SLOT + tid] = v;
    s_slot[tid] = v;
  }
  __syncthreads();
  if (tid < 192) s_qv[tid] = compute_q(p, s_slot, tid >> 6, tid & 63);
  __syncthreads();
  if (tid < 192) p.ws[WS_Q + tid] = compute_qp(p, s_qv, tid >> 6, tid & 63);
}

// ---------------------------------------------------------------------------
// Streaming pass (thread-per-row): LN in registers, per-thread logits with
// Q', per-thread softmax (no cross-lane), coalesced attn writes, and the
// y(3x64) += w^T(3x32) @ xn(32x64) rank update done with one
// v_wmma_f32_16x16x32_f16 per 16-column tile, accumulators held in VGPRs.
// ---------------------------------------------------------------------------
__global__ __launch_bounds__(256) void slot_pass_k(Params p, int write_attn) {
  __shared__ __align__(16) _Float16 s_xc[8 * 64 * 40]; // [wave][col][row(32)+pad]
  __shared__ __align__(16) _Float16 s_wt[8 * 4 * 40];  // [wave][s(3+pad)][row(32)+pad]
  __shared__ __align__(16) float s_qp[192];
  __shared__ __align__(8) float2 s_gb[64];             // {ln_g, ln_b} packed
  __shared__ float s_pn[8 * 3 * 64];                   // per-wave y partials
  __shared__ float s_pd[8 * 3];                        // per-wave den partials

  const int tid  = threadIdx.x;
  const int wave = tid >> 5;
  const int lane = tid & 31;
  const int ln16 = lane & 15;
  const int hlf  = lane >> 4;

  // ---- stage Q', LN params; zero the w^T strip (incl. padding row) --------
  if (tid < 192) s_qp[tid] = p.ws[WS_Q + tid];
  if (tid < 64) { s_gb[tid] = make_float2(p.ln_in_g[tid], p.ln_in_b[tid]); }
  for (int i = tid; i < (8 * 4 * 40) / 2; i += 256) ((unsigned*)s_wt)[i] = 0u;
  __syncthreads();

  float denAcc[3] = {0.f, 0.f, 0.f};
  const v8f z8 = {0.f, 0.f, 0.f, 0.f, 0.f, 0.f, 0.f, 0.f};
  v8f yacc[4] = {z8, z8, z8, z8};

  float* out_attn = p.out + 192;
  _Float16* xcw = &s_xc[wave * 64 * 40];   // this wave's column-major xn
  _Float16* wtw = &s_wt[wave * 4 * 40];    // this wave's w^T strip

  #pragma unroll 1
  for (int c = 0; c < CHUNKS; ++c) {
    asm volatile("" ::: "memory");   // keep Q'/param LDS reads inside the loop
    const long rowBase = ((long)blockIdx.x * CHUNKS + c) * 256;

    // ---- load own row, LayerNorm (keep f32 in regs, f16 col-major in LDS) --
    float xv[64];
    {
      const float4* xr = (const float4*)(p.x + (rowBase + tid) * 64);
      #pragma unroll
      for (int q4 = 0; q4 < 16; ++q4) {
        float4 t = xr[q4];
        xv[4 * q4 + 0] = t.x; xv[4 * q4 + 1] = t.y;
        xv[4 * q4 + 2] = t.z; xv[4 * q4 + 3] = t.w;
      }
      // 4-way tree accumulation: short dependency chains
      float m0 = 0.f, m1 = 0.f, m2 = 0.f, m3 = 0.f;
      #pragma unroll
      for (int k4 = 0; k4 < 16; ++k4) {
        m0 += xv[4 * k4 + 0]; m1 += xv[4 * k4 + 1];
        m2 += xv[4 * k4 + 2]; m3 += xv[4 * k4 + 3];
      }
      float mean = ((m0 + m1) + (m2 + m3)) * (1.f / 64.f);
      float v0 = 0.f, v1 = 0.f, v2 = 0.f, v3 = 0.f;
      #pragma unroll
      for (int k4 = 0; k4 < 16; ++k4) {
        float d0 = xv[4 * k4 + 0] - mean, d1 = xv[4 * k4 + 1] - mean;
        float d2 = xv[4 * k4 + 2] - mean, d3 = xv[4 * k4 + 3] - mean;
        v0 += d0 * d0; v1 += d1 * d1; v2 += d2 * d2; v3 += d3 * d3;
      }
      float var = ((v0 + v1) + (v2 + v3)) * (1.f / 64.f);
      float inv = rsqrtf(var + 1e-5f);
      #pragma unroll
      for (int k = 0; k < 64; ++k) {
        float2 gb = s_gb[k];
        float xn = (xv[k] - mean) * inv * gb.x + gb.y;
        xv[k] = xn;
        xcw[k * 40 + lane] = (_Float16)xn;   // column-major for WMMA B frags
      }
    }

    // ---- per-thread logits: l[s] = xn . Q'[s] ------------------------------
    float l0 = 0.f, l1 = 0.f, l2 = 0.f;
    #pragma unroll
    for (int k4 = 0; k4 < 16; ++k4) {
      float4 q0 = *(const float4*)&s_qp[0 * 64 + 4 * k4];
      float4 q1 = *(const float4*)&s_qp[1 * 64 + 4 * k4];
      float4 q2 = *(const float4*)&s_qp[2 * 64 + 4 * k4];
      float x0 = xv[4 * k4 + 0], x1 = xv[4 * k4 + 1];
      float x2 = xv[4 * k4 + 2], x3 = xv[4 * k4 + 3];
      l0 += x0 * q0.x + x1 * q0.y + x2 * q0.z + x3 * q0.w;
      l1 += x0 * q1.x + x1 * q1.y + x2 * q1.z + x3 * q1.w;
      l2 += x0 * q2.x + x1 * q2.y + x2 * q2.z + x3 * q2.w;
    }

    // ---- per-thread softmax, coalesced attn store, den ---------------------
    float mx = fmaxf(l0, fmaxf(l1, l2));
    float e0 = __expf(l0 - mx), e1 = __expf(l1 - mx), e2 = __expf(l2 - mx);
    float rs = 1.f / (e0 + e1 + e2);
    float a0v = e0 * rs, a1v = e1 * rs, a2v = e2 * rs;
    if (write_attn) {
      long rg = rowBase + tid;
      out_attn[0 * (long)N_TOK + rg] = a0v;
      out_attn[1 * (long)N_TOK + rg] = a1v;
      out_attn[2 * (long)N_TOK + rg] = a2v;
    }
    float w0 = a0v + EPSV, w1 = a1v + EPSV, w2 = a2v + EPSV;
    denAcc[0] += w0; denAcc[1] += w1; denAcc[2] += w2;

    // ---- stage w^T (3 x 32 f16), then y += w^T @ xn via WMMA ---------------
    wtw[0 * 40 + lane] = (_Float16)w0;
    wtw[1 * 40 + lane] = (_Float16)w1;
    wtw[2 * 40 + lane] = (_Float16)w2;

    // A fragment: 16x32 f16 layout; lane row m = ln16 (only m<3 meaningful,
    // rows 3..15 alias rows 0..3 of the zero-padded strip -> harmless)
    VH aw;
    const _Float16* ab = &s_wt[wave * 160 + (ln16 & 3) * 40 + hlf * 8];
    aw.q[0] = *(const uint4*)(ab);
    aw.q[1] = *(const uint4*)(ab + 16);

    #pragma unroll
    for (int t = 0; t < 4; ++t) {
      // B fragment: 32x16 f16, K (=row) in table order (8h+s, 16+8h+s-8)
      VH bx;
      const _Float16* bb = &s_xc[(wave * 64 + t * 16 + ln16) * 40 + hlf * 8];
      bx.q[0] = *(const uint4*)(bb);
      bx.q[1] = *(const uint4*)(bb + 16);
      yacc[t] = __builtin_amdgcn_wmma_f32_16x16x32_f16(
          false, aw.v, false, bx.v, (short)0, yacc[t], false, false);
    }
  }

  // ---- publish per-wave partials ------------------------------------------
  // y: C/D layout -> lane (hlf==0) VGPR j holds y[s=j][t*16 + ln16]
  if (hlf == 0) {
    #pragma unroll
    for (int t = 0; t < 4; ++t) {
      s_pn[(wave * 3 + 0) * 64 + t * 16 + ln16] = yacc[t][0];
      s_pn[(wave * 3 + 1) * 64 + t * 16 + ln16] = yacc[t][1];
      s_pn[(wave * 3 + 2) * 64 + t * 16 + ln16] = yacc[t][2];
    }
  }
  // den: one full-wave butterfly at kernel end (deterministic)
  #pragma unroll
  for (int mask = 1; mask < 32; mask <<= 1) {
    denAcc[0] += __shfl_xor(denAcc[0], mask, 32);
    denAcc[1] += __shfl_xor(denAcc[1], mask, 32);
    denAcc[2] += __shfl_xor(denAcc[2], mask, 32);
  }
  if (lane == 0) {
    s_pd[wave * 3 + 0] = denAcc[0];
    s_pd[wave * 3 + 1] = denAcc[1];
    s_pd[wave * 3 + 2] = denAcc[2];
  }
  __syncthreads();

  float* part = p.ws + WS_PART + (long)blockIdx.x * 208;
  if (tid < 192) {
    int s = tid >> 6, k = tid & 63;
    float acc = 0.f;
    #pragma unroll
    for (int w = 0; w < 8; ++w) acc += s_pn[(w * 3 + s) * 64 + k];
    part[tid] = acc;
  }
  if (tid < 3) {
    float acc = 0.f;
    #pragma unroll
    for (int w = 0; w < 8; ++w) acc += s_pd[w * 3 + tid];
    part[192 + tid] = acc;
  }
}

// ---------------------------------------------------------------------------
// Slot update: reduce partials, num = y @ Wv.T, GRU, MLP, next Q' / output
// ---------------------------------------------------------------------------
__global__ __launch_bounds__(256) void slot_update_k(Params p, int last) {
  __shared__ float s_red[208];    // y (192) + den (3)
  __shared__ float s_upd[192];
  __shared__ float s_h[192];
  __shared__ float s_new[192];
  __shared__ float s_ln[192];
  __shared__ float s_hid[384];
  __shared__ float s_qv[192];
  const int tid = threadIdx.x;
  const int s = tid >> 6, d = tid & 63;

  // deterministic fixed-order reduction over NBLK_P block partials
  for (int i = tid; i < 195; i += 256) {
    float acc = 0.f;
    const float* base = p.ws + WS_PART + i;
    #pragma unroll 4
    for (int b = 0; b < NBLK_P; ++b) acc += base[(long)b * 208];
    s_red[i] = acc;
  }
  __syncthreads();

  if (tid < 192) {
    // num[s][d] = sum_k y[s][k] * Wv[d][k] ;  upd = num / den
    const float* y = s_red + s * 64;
    const float* w = p.Wv + d * 64;
    float acc = 0.f;
    #pragma unroll 8
    for (int k = 0; k < 64; ++k) acc += y[k] * w[k];
    s_upd[tid] = acc / s_red[192 + s];
    s_h[tid]   = p.ws[WS_SLOT + tid];
  }
  __syncthreads();

  float hnew = 0.f;
  if (tid < 192) {
    const float* wih = (s == 0) ? p.gbg_wih : p.gfg_wih;
    const float* whh = (s == 0) ? p.gbg_whh : p.gfg_whh;
    const float* bih = (s == 0) ? p.gbg_bih : p.gfg_bih;
    const float* bhh = (s == 0) ? p.gbg_bhh : p.gfg_bhh;
    const float* xu = s_upd + s * 64;
    const float* hh = s_h + s * 64;
    float gi[3], gh[3];
    #pragma unroll
    for (int g = 0; g < 3; ++g) {
      int row = g * 64 + d;
      float ai = bih[row], ah = bhh[row];
      const float* wi = wih + row * 64;
      const float* wh = whh + row * 64;
      #pragma unroll 8
      for (int k = 0; k < 64; ++k) { ai += xu[k] * wi[k]; ah += hh[k] * wh[k]; }
      gi[g] = ai; gh[g] = ah;
    }
    float r = 1.f / (1.f + __expf(-(gi[0] + gh[0])));
    float z = 1.f / (1.f + __expf(-(gi[1] + gh[1])));
    float n = tanhf(gi[2] + r * gh[2]);
    hnew = (1.f - z) * n + z * hh[d];
    s_new[tid] = hnew;
  }
  __syncthreads();

  if (tid < 192) {      // MLP layernorm
    const float* g = (s == 0) ? p.mbg_ln_g : p.mfg_ln_g;
    const float* b = (s == 0) ? p.mbg_ln_b : p.mfg_ln_b;
    const float* sl = s_new + s * 64;
    float m = 0.f;
    #pragma unroll
    for (int k = 0; k < 64; ++k) m += sl[k];
    m *= (1.f / 64.f);
    float var = 0.f;
    #pragma unroll
    for (int k = 0; k < 64; ++k) { float t = sl[k] - m; var += t * t; }
    var *= (1.f / 64.f);
    float inv = rsqrtf(var + 1e-5f);
    s_ln[tid] = (sl[d] - m) * inv * g[d] + b[d];
  }
  __syncthreads();

  if (tid < 192) {      // hidden layer (128 per slot, 2 per thread)
    const float* w1 = (s == 0) ? p.mbg_w1 : p.mfg_w1;
    const float* b1 = (s == 0) ? p.mbg_b1 : p.mfg_b1;
    const float* lnv = s_ln + s * 64;
    #pragma unroll
    for (int h2 = 0; h2 < 2; ++h2) {
      int j = h2 * 64 + d;
      float acc = b1[j];
      const float* w = w1 + j * 64;
      #pragma unroll 8
      for (int k = 0; k < 64; ++k) acc += lnv[k] * w[k];
      s_hid[s * 128 + j] = fmaxf(acc, 0.f);
    }
  }
  __syncthreads();

  if (tid < 192) {      // output projection + residual
    const float* w2 = (s == 0) ? p.mbg_w2 : p.mfg_w2;
    const float* b2 = (s == 0) ? p.mbg_b2 : p.mfg_b2;
    const float* hid = s_hid + s * 128;
    float acc = b2[d];
    const float* w = w2 + d * 128;
    #pragma unroll 8
    for (int k = 0; k < 128; ++k) acc += hid[k] * w[k];
    float outv = hnew + acc;
    p.ws[WS_SLOT + tid] = outv;
    s_new[tid] = outv;
    if (last) p.out[tid] = outv;
  }
  __syncthreads();
  if (!last && tid < 192) s_qv[tid] = compute_q(p, s_new, s, d);
  __syncthreads();
  if (!last && tid < 192) p.ws[WS_Q + tid] = compute_qp(p, s_qv, s, d);
}

// ---------------------------------------------------------------------------
extern "C" void kernel_launch(void* const* d_in, const int* in_sizes, int n_in,
                              void* d_out, int out_size, void* d_ws, size_t ws_size,
                              hipStream_t stream) {
  (void)in_sizes; (void)n_in; (void)out_size; (void)ws_size;
  Params p;
  p.x         = (const float*)d_in[0];
  p.noise_fg  = (const float*)d_in[1];
  p.noise_bg  = (const float*)d_in[2];
  p.ln_in_g   = (const float*)d_in[3];
  p.ln_in_b   = (const float*)d_in[4];
  p.mu_fg     = (const float*)d_in[5];
  p.logsig_fg = (const float*)d_in[6];
  p.mu_bg     = (const float*)d_in[7];
  p.logsig_bg = (const float*)d_in[8];
  p.Wk        = (const float*)d_in[9];
  p.Wv        = (const float*)d_in[10];
  p.lnq_fg_g  = (const float*)d_in[11];
  p.lnq_fg_b  = (const float*)d_in[12];
  p.Wq_fg     = (const float*)d_in[13];
  p.lnq_bg_g  = (const float*)d_in[14];
  p.lnq_bg_b  = (const float*)d_in[15];
  p.Wq_bg     = (const float*)d_in[16];
  p.gfg_wih   = (const float*)d_in[17];
  p.gfg_whh   = (const float*)d_in[18];
  p.gfg_bih   = (const float*)d_in[19];
  p.gfg_bhh   = (const float*)d_in[20];
  p.gbg_wih   = (const float*)d_in[21];
  p.gbg_whh   = (const float*)d_in[22];
  p.gbg_bih   = (const float*)d_in[23];
  p.gbg_bhh   = (const float*)d_in[24];
  p.mfg_ln_g  = (const float*)d_in[25];
  p.mfg_ln_b  = (const float*)d_in[26];
  p.mfg_w1    = (const float*)d_in[27];
  p.mfg_b1    = (const float*)d_in[28];
  p.mfg_w2    = (const float*)d_in[29];
  p.mfg_b2    = (const float*)d_in[30];
  p.mbg_ln_g  = (const float*)d_in[31];
  p.mbg_ln_b  = (const float*)d_in[32];
  p.mbg_w1    = (const float*)d_in[33];
  p.mbg_b1    = (const float*)d_in[34];
  p.mbg_w2    = (const float*)d_in[35];
  p.mbg_b2    = (const float*)d_in[36];
  p.out = (float*)d_out;
  p.ws  = (float*)d_ws;

  slot_init_k  <<<1,      256, 0, stream>>>(p);
  slot_pass_k  <<<NBLK_P, 256, 0, stream>>>(p, 0);   // iteration 1
  slot_update_k<<<1,      256, 0, stream>>>(p, 0);
  slot_pass_k  <<<NBLK_P, 256, 0, stream>>>(p, 1);   // iteration 2 (+ attn out)
  slot_update_k<<<1,      256, 0, stream>>>(p, 1);   // final slots
}
